// GroupedQueryAttention_3624952398460
// MI455X (gfx1250) — matrix operand order repaired and checked by reading, compiled, not run on hardware
//
#include <hip/hip_runtime.h>

// ---------------------------------------------------------------------------
// GQA forward for MI455X (gfx1250): bf16 WMMA + async global->LDS pipeline.
// ---------------------------------------------------------------------------

typedef __bf16 bf16;
typedef bf16  v16bf __attribute__((ext_vector_type(16)));
typedef bf16  v8bf  __attribute__((ext_vector_type(8)));
typedef float v8f   __attribute__((ext_vector_type(8)));

#define DIMC   4096
#define NH     32
#define NKV    8
#define HD     128
#define BSZ    2
#define TSZ    2048
#define MROWS  (BSZ * TSZ)      // 4096 token rows
#define KVC    (2 * NKV * HD)   // 2048 kv-projection cols
#define LDSW   40               // padded LDS row stride (32 data + 8 pad bf16)

static __device__ __forceinline__ v8f wmma_bf16(v16bf a, v16bf b, v8f c) {
  // D = A(16x32 bf16) * B(32x16 bf16) + C(16x16 f32)
  return __builtin_amdgcn_wmma_f32_16x16x32_bf16(
      /*neg_a=*/false, a, /*neg_b=*/false, b,
      /*c_mod=*/(short)0, c, /*reuse_a=*/false, /*reuse_b=*/false);
}

static __device__ __forceinline__ v16bf combine8(v8bf lo, v8bf hi) {
  v16bf r;
#pragma unroll
  for (int i = 0; i < 8; ++i) { r[i] = lo[i]; r[8 + i] = hi[i]; }
  return r;
}

// 16-byte async copy global -> LDS (ASYNCcnt-tracked; bypasses VGPRs).
// Low 32 bits of a generic pointer to a __shared__ object are its LDS offset.
static __device__ __forceinline__ void async_copy16(const bf16* lds_dst,
                                                    const bf16* gsrc) {
  unsigned off = (unsigned)(uintptr_t)lds_dst;
  asm volatile("global_load_async_to_lds_b128 %0, %1, off"
               :: "v"(off), "v"(gsrc)
               : "memory");
}

static __device__ __forceinline__ void wait_async0() {
  asm volatile("s_wait_asynccnt 0" ::: "memory");
}

// ---------------------------------------------------------------------------
// fp32 -> bf16 elementwise convert (grid-stride)
// ---------------------------------------------------------------------------
__global__ void cvt_f32_to_bf16(const float* __restrict__ src,
                                bf16* __restrict__ dst, long long n) {
  long long i = (long long)blockIdx.x * blockDim.x + threadIdx.x;
  long long stride = (long long)gridDim.x * blockDim.x;
  for (; i < n; i += stride) dst[i] = (bf16)src[i];
}

// ---------------------------------------------------------------------------
// Build Vt[b, g, d, t] = V[b, t, g, d]  (V = second half of kv projection)
// ---------------------------------------------------------------------------
__global__ void build_vt(const bf16* __restrict__ kvb, bf16* __restrict__ vt) {
  long long idx = (long long)blockIdx.x * blockDim.x + threadIdx.x;
  const long long total = (long long)BSZ * NKV * HD * TSZ;
  long long stride = (long long)gridDim.x * blockDim.x;
  for (; idx < total; idx += stride) {
    int t = (int)(idx % TSZ);
    long long r = idx / TSZ;
    int d = (int)(r % HD); r /= HD;
    int g = (int)(r % NKV);
    int b = (int)(r / NKV);
    vt[idx] = kvb[((long long)(b * TSZ + t)) * KVC + NKV * HD + g * HD + d];
  }
}

// ---------------------------------------------------------------------------
// C[M,N] = A[M,K] * B[N,K]^T  (bf16 in, f32 accumulate).
// Block: 256 thr = 8 waves, tile 128(M) x 128(N); wave: 16(M) x 128(N).
// K staged in 32-wide double-buffered LDS tiles filled by the async engine
// (global_load_async_to_lds_b128), overlapped with WMMA on the other buffer.
// An asm materialization fence forces all 18 DS fragment loads of a k-step
// to complete under ONE s_wait_dscnt, then the 8 WMMAs issue back-to-back.
// ---------------------------------------------------------------------------
__global__ __launch_bounds__(256) void gemm_bf16_tn(
    const bf16* __restrict__ A, const bf16* __restrict__ Bm,
    bf16* __restrict__ Cb, float* __restrict__ Cf,
    int M, int N, int K, int writeF32) {
  __shared__ __align__(16) bf16 As[2][128 * LDSW];   // 2 x 10 KB
  __shared__ __align__(16) bf16 Bs[2][128 * LDSW];   // 2 x 10 KB

  const int tid  = threadIdx.x;
  const int lane = tid & 31;
  const int wave = tid >> 5;
  const int l15  = lane & 15;
  const int lh   = lane >> 4;
  const int mblk = blockIdx.y * 128;
  const int nblk = blockIdx.x * 128;

  // staging pattern: 512 16B chunks per 128x32 tile -> 2 chunks/thread/tile
  const int srow = tid >> 2;
  const int ssub = (tid & 3) * 8;

  auto stage_tile = [&](int buf, int k0) {
    async_copy16(&As[buf][srow * LDSW + ssub],
                 A + (size_t)(mblk + srow) * K + k0 + ssub);
    async_copy16(&As[buf][(srow + 64) * LDSW + ssub],
                 A + (size_t)(mblk + srow + 64) * K + k0 + ssub);
    async_copy16(&Bs[buf][srow * LDSW + ssub],
                 Bm + (size_t)(nblk + srow) * K + k0 + ssub);
    async_copy16(&Bs[buf][(srow + 64) * LDSW + ssub],
                 Bm + (size_t)(nblk + srow + 64) * K + k0 + ssub);
  };

  v8f acc[8];
#pragma unroll
  for (int i = 0; i < 8; ++i)
#pragma unroll
    for (int r = 0; r < 8; ++r) acc[i][r] = 0.0f;

  stage_tile(0, 0);                       // prologue fill

  const int KT = K >> 5;                  // K/32 tiles
  for (int t = 0; t < KT; ++t) {
    const int buf = t & 1;
    wait_async0();                        // this wave's staged chunks landed
    __syncthreads();                      // every wave's chunks landed,
                                          // and buf^1 is no longer being read
    if (t + 1 < KT) stage_tile(buf ^ 1, (t + 1) * 32);

    // ---- load all fragments for this k-step ----
    // A fragment: wave row strip; 16 contiguous K at lh*16
    const bf16* ab = &As[buf][(wave * 16 + l15) * LDSW + lh * 16];
    v16bf af = combine8(*(const v8bf*)ab, *(const v8bf*)(ab + 8));
    v16bf bfr[8];
#pragma unroll
    for (int i = 0; i < 8; ++i) {
      // B fragment: col i*16+l15; 16 contiguous K at lh*16
      const bf16* bb = &Bs[buf][(i * 16 + l15) * LDSW + lh * 16];
      bfr[i] = combine8(*(const v8bf*)bb, *(const v8bf*)(bb + 8));
    }
    // materialization fence: all DS loads complete (single s_wait_dscnt)
    // before the WMMA chain; stops the scheduler re-sinking loads per WMMA.
    asm volatile("" : "+v"(af),
                      "+v"(bfr[0]), "+v"(bfr[1]), "+v"(bfr[2]), "+v"(bfr[3]),
                      "+v"(bfr[4]), "+v"(bfr[5]), "+v"(bfr[6]), "+v"(bfr[7]));
    // ---- back-to-back WMMA chain (independent accumulators) ----
#pragma unroll
    for (int i = 0; i < 8; ++i) acc[i] = wmma_bf16(af, bfr[i], acc[i]);
  }

#pragma unroll
  for (int i = 0; i < 8; ++i)
#pragma unroll
    for (int r = 0; r < 8; ++r) {
      int row = mblk + wave * 16 + r + lh * 8;  // C/D: VGPR r -> M=r+8*(lane>=16)
      int col = nblk + i * 16 + l15;
      if (writeF32) Cf[(size_t)row * N + col] = acc[i][r];
      else          Cb[(size_t)row * N + col] = (bf16)acc[i][r];
    }
}

// ---------------------------------------------------------------------------
// Flash GQA: one wave per 16-query tile, 32-key blocks, online softmax.
// Q rows from q_b [MROWS, DIM]; K rows from kv_b [MROWS, KVC];
// V^T from vt [B,KV,HD,T]. Output o_b [MROWS, DIM] bf16.
// ---------------------------------------------------------------------------
__global__ __launch_bounds__(128) void gqa_flash(
    const bf16* __restrict__ Qb, const bf16* __restrict__ KVb,
    const bf16* __restrict__ Vt, bf16* __restrict__ Ob) {
  __shared__ __align__(16) bf16 lds[4][16][32];   // per-wave P transpose tile

  const int lane  = threadIdx.x & 31;
  const int wave  = threadIdx.x >> 5;
  const int l15   = lane & 15;
  const int lh    = lane >> 4;
  const int mbase = blockIdx.x * 64 + wave * 16;  // query tile start
  const int bh    = blockIdx.y;                   // 0..B*NH-1
  const int b     = bh >> 5;
  const int h     = bh & 31;
  const int g     = h >> 2;                       // kv head (R=4)
  const float inv_scale = 0.08838834764831845f;   // 1/sqrt(128)

  // Q fragments: hd=128 -> 4 K=32 chunks, kept resident
  v16bf qf[4];
  {
    const bf16* qrow = Qb + (size_t)(b * TSZ + mbase + l15) * DIMC + h * HD + lh * 16;
#pragma unroll
    for (int kk = 0; kk < 4; ++kk) {
      v8bf lo = *(const v8bf*)(qrow + kk * 32);
      v8bf hi = *(const v8bf*)(qrow + kk * 32 + 8);
      qf[kk] = combine8(lo, hi);
    }
  }

  v8f accO[8];                       // 16 x 128 output tile
#pragma unroll
  for (int g2 = 0; g2 < 8; ++g2)
#pragma unroll
    for (int r = 0; r < 8; ++r) accO[g2][r] = 0.0f;

  float mrow[8], lrow[8];
#pragma unroll
  for (int r = 0; r < 8; ++r) { mrow[r] = -1.0e30f; lrow[r] = 0.0f; }

  const bf16* kbase = KVb + (size_t)b * TSZ * KVC + g * HD;
  const bf16* vbase = Vt + (size_t)(b * NKV + g) * HD * TSZ;

  const int mEnd = mbase + 15;
  for (int s = 0; s <= mEnd; s += 32) {
    // ---- S = Q K^T  (16 x 32, two f32 accumulators) ----
    v8f s0, s1;
#pragma unroll
    for (int r = 0; r < 8; ++r) { s0[r] = 0.0f; s1[r] = 0.0f; }
#pragma unroll
    for (int kk = 0; kk < 4; ++kk) {
      const bf16* k0 = kbase + (size_t)(s + l15) * KVC + kk * 32 + lh * 16;
      const bf16* k1 = kbase + (size_t)(s + 16 + l15) * KVC + kk * 32 + lh * 16;
      v16bf bf0 = *(const v16bf*)k0;
      v16bf bf1 = *(const v16bf*)k1;
      s0 = wmma_bf16(qf[kk], bf0, s0);
      s1 = wmma_bf16(qf[kk], bf1, s1);
    }

    const bool need_mask = (s + 31 > mbase);
#pragma unroll
    for (int r = 0; r < 8; ++r) {
      float x0 = s0[r] * inv_scale;
      float x1 = s1[r] * inv_scale;
      if (need_mask) {
        int mg = mbase + r + lh * 8;
        if (s + l15 > mg)      x0 = -1.0e30f;
        if (s + 16 + l15 > mg) x1 = -1.0e30f;
      }
      s0[r] = x0; s1[r] = x1;
    }

    // ---- online softmax (row reductions across 16-lane groups) ----
    float alpha[8];
#pragma unroll
    for (int r = 0; r < 8; ++r) {
      float mx = fmaxf(s0[r], s1[r]);
      mx = fmaxf(mx, __shfl_xor(mx, 1, 32));
      mx = fmaxf(mx, __shfl_xor(mx, 2, 32));
      mx = fmaxf(mx, __shfl_xor(mx, 4, 32));
      mx = fmaxf(mx, __shfl_xor(mx, 8, 32));
      float mnew = fmaxf(mrow[r], mx);
      alpha[r] = __expf(mrow[r] - mnew);
      mrow[r] = mnew;
      float p0 = __expf(s0[r] - mnew);
      float p1 = __expf(s1[r] - mnew);
      s0[r] = p0; s1[r] = p1;
      float ps = p0 + p1;
      ps += __shfl_xor(ps, 1, 32);
      ps += __shfl_xor(ps, 2, 32);
      ps += __shfl_xor(ps, 4, 32);
      ps += __shfl_xor(ps, 8, 32);
      lrow[r] = lrow[r] * alpha[r] + ps;
    }

    // ---- transpose P (C/D layout) -> A-fragment layout via per-wave LDS ----
#pragma unroll
    for (int r = 0; r < 8; ++r) {
      int m = r + lh * 8;
      lds[wave][m][l15]      = (bf16)s0[r];
      lds[wave][m][16 + l15] = (bf16)s1[r];
    }
    v16bf pf;
    {
      // same-wave DS ops are in-order: read back safely without a barrier
      v8bf lo = *(const v8bf*)&lds[wave][l15][lh * 16];
      v8bf hi = *(const v8bf*)&lds[wave][l15][lh * 16 + 8];
      pf = combine8(lo, hi);
    }

    // ---- rescale accumulator, then O += P * V ----
#pragma unroll
    for (int g2 = 0; g2 < 8; ++g2)
#pragma unroll
      for (int r = 0; r < 8; ++r) accO[g2][r] *= alpha[r];

#pragma unroll
    for (int g2 = 0; g2 < 8; ++g2) {
      const bf16* vrow = vbase + (size_t)(g2 * 16 + l15) * TSZ + s + lh * 16;
      v16bf vf = *(const v16bf*)vrow;
      accO[g2] = wmma_bf16(pf, vf, accO[g2]);
    }
  }

  // ---- epilogue: normalize, store bf16 in [b,t,h,d] flat layout ----
#pragma unroll
  for (int r = 0; r < 8; ++r) lrow[r] = 1.0f / lrow[r];
#pragma unroll
  for (int g2 = 0; g2 < 8; ++g2)
#pragma unroll
    for (int r = 0; r < 8; ++r) {
      size_t row = (size_t)(b * TSZ + mbase + r + lh * 8);
      Ob[row * DIMC + h * HD + g2 * 16 + l15] = (bf16)(accO[g2][r] * lrow[r]);
    }
}

// ---------------------------------------------------------------------------
// Host-side launch
// ---------------------------------------------------------------------------
extern "C" void kernel_launch(void* const* d_in, const int* in_sizes, int n_in,
                              void* d_out, int out_size, void* d_ws, size_t ws_size,
                              hipStream_t stream) {
  (void)in_sizes; (void)n_in; (void)out_size; (void)ws_size;
  const float* x   = (const float*)d_in[0];
  const float* Wq  = (const float*)d_in[1];
  const float* Wkv = (const float*)d_in[2];
  const float* Wo  = (const float*)d_in[3];
  float* out = (float*)d_out;

  // scratch layout (bytes)
  char* ws = (char*)d_ws;
  const size_t SZ_X   = (size_t)MROWS * DIMC * sizeof(bf16);  // 32 MB
  const size_t SZ_WQ  = (size_t)DIMC * DIMC * sizeof(bf16);   // 32 MB
  const size_t SZ_WKV = (size_t)KVC * DIMC * sizeof(bf16);    // 16 MB
  const size_t SZ_WO  = SZ_WQ;                                // 32 MB
  const size_t SZ_Q   = SZ_X;                                 // 32 MB
  const size_t SZ_KV  = (size_t)MROWS * KVC * sizeof(bf16);   // 16 MB
  const size_t SZ_VT  = (size_t)BSZ * NKV * HD * TSZ * sizeof(bf16); // 8 MB

  size_t off = 0;
  bf16* xb   = (bf16*)(ws + off); off += SZ_X;
  bf16* wqb  = (bf16*)(ws + off); off += SZ_WQ;
  bf16* wkvb = (bf16*)(ws + off); off += SZ_WKV;
  bf16* wob  = (bf16*)(ws + off); off += SZ_WO;
  bf16* qb   = (bf16*)(ws + off); off += SZ_Q;
  bf16* kvb  = (bf16*)(ws + off); off += SZ_KV;
  bf16* vtb  = (bf16*)(ws + off); off += SZ_VT;
  bf16* ob   = (bf16*)(ws + off);

  // 1) fp32 -> bf16 converts
  cvt_f32_to_bf16<<<4096, 256, 0, stream>>>(x,   xb,  (long long)MROWS * DIMC);
  cvt_f32_to_bf16<<<4096, 256, 0, stream>>>(Wq,  wqb, (long long)DIMC * DIMC);
  cvt_f32_to_bf16<<<2048, 256, 0, stream>>>(Wkv, wkvb,(long long)KVC * DIMC);
  cvt_f32_to_bf16<<<4096, 256, 0, stream>>>(Wo,  wob, (long long)DIMC * DIMC);

  // 2) Q = x Wq^T   [4096 x 4096]
  gemm_bf16_tn<<<dim3(DIMC / 128, MROWS / 128), 256, 0, stream>>>(
      xb, wqb, qb, nullptr, MROWS, DIMC, DIMC, 0);

  // 3) KV = x Wkv^T [4096 x 2048]
  gemm_bf16_tn<<<dim3(KVC / 128, MROWS / 128), 256, 0, stream>>>(
      xb, wkvb, kvb, nullptr, MROWS, KVC, DIMC, 0);

  // 4) V^T staging for P*V fragments
  build_vt<<<2048, 256, 0, stream>>>(kvb, vtb);

  // 5) flash GQA: grid (T/64 query tiles, B*NH heads), 4 waves/block
  gqa_flash<<<dim3(TSZ / 64, BSZ * NH), 128, 0, stream>>>(qb, kvb, vtb, ob);

  // 6) out = attn_out Wo^T (fp32 result)
  gemm_bf16_tn<<<dim3(DIMC / 128, MROWS / 128), 256, 0, stream>>>(
      ob, wob, nullptr, out, MROWS, DIMC, DIMC, 1);
}
